// LinearTemporalDiffusionTransformerDecoderLayer_70660801954591
// MI455X (gfx1250) — matrix operand order repaired and checked by reading, not compiled, not run on hardware
//
// CDNA5 (gfx1250) implementation of LinearTemporalDiffusionTransformerDecoderLayer.
// All GEMMs run on v_wmma_f32_16x16x32_bf16 (f32 -> bf16 staged through LDS,
// f32 accumulate). Row-wise ops (LN, softmaxes, FiLM+SiLU) are elementwise kernels.
#include <hip/hip_runtime.h>

// ---------------------------------------------------------------- vector types
typedef __attribute__((ext_vector_type(16))) __bf16       bf16x16;
typedef __attribute__((ext_vector_type(8)))  float        floatx8;
typedef __attribute__((ext_vector_type(4)))  float        f32x4;
typedef __attribute__((ext_vector_type(4)))  unsigned int u32x4;
typedef __attribute__((ext_vector_type(2)))  unsigned int u32x2;

union FragBF { bf16x16 v; u32x4 q[2]; };

// ---------------------------------------------------------------- constants
#define B_   8
#define T_   2048
#define NN_  256
#define D_   1024
#define H_   16
#define DH_  64
#define TE_  2048
#define FD_  4096

#define FLAG_TA      1
#define FLAG_TB      2
#define FLAG_GELU    4
#define FLAG_SILU    8
#define FLAG_PSILU   16

// ---------------------------------------------------------------- device utils
__device__ __forceinline__ unsigned short f2bf(float f) {
  unsigned int u = __float_as_uint(f);
  u += 0x7FFFu + ((u >> 16) & 1u);       // round-to-nearest-even
  return (unsigned short)(u >> 16);
}
__device__ __forceinline__ unsigned int f2bf_pk(float lo, float hi) {
  return (unsigned int)f2bf(lo) | ((unsigned int)f2bf(hi) << 16);
}
__device__ __forceinline__ float silu_f(float x) { return x / (1.f + expf(-x)); }
__device__ __forceinline__ float gelu_f(float x) {
  return 0.5f * x * (1.f + erff(x * 0.70710678118654752440f));
}

// ---------------------------------------------------------------- GEMM staging
// Stage a [OUT x 32] bf16 tile into LDS (stride 40 shorts). contig_k selects
// which memory direction is contiguous. Out-of-range elements staged as zero.
template <int OUT, int NTHR>
__device__ __forceinline__ void stage_tile(unsigned short* sm, const float* __restrict__ p,
                                           int ld, bool contig_k, int out0, int k0,
                                           int OUTmax, int Kmax, int tid, bool presilu) {
  const int LSTR = 40;
  if (contig_k) {
    for (int s = tid; s < OUT * 8; s += NTHR) {
      int o = s >> 3, k4 = (s & 7) << 2;
      int go = out0 + o, gk = k0 + k4;
      f32x4 v; v[0] = v[1] = v[2] = v[3] = 0.f;
      if (go < OUTmax) {
        const float* g = p + (long long)go * ld + gk;
        if (gk + 3 < Kmax) {
          v = *(const f32x4*)g;
          if (gk + 35 < Kmax) __builtin_prefetch(g + 32, 0, 1);   // global_prefetch
        } else {
          for (int i = 0; i < 4; ++i) if (gk + i < Kmax) v[i] = g[i];
        }
      }
      if (presilu) {
#pragma unroll
        for (int i = 0; i < 4; ++i) v[i] = silu_f(v[i]);
      }
      // 4 bf16 = 8 bytes, 8B-aligned (row stride 80B, k4*2 multiple of 8)
      u32x2 pk; pk[0] = f2bf_pk(v[0], v[1]); pk[1] = f2bf_pk(v[2], v[3]);
      *(u32x2*)(sm + o * LSTR + k4) = pk;
    }
  } else {
    const int O4 = OUT / 4;
    for (int s = tid; s < OUT * 8; s += NTHR) {
      int k = s / O4, o4 = (s % O4) << 2;
      int gk = k0 + k, go = out0 + o4;
      f32x4 v; v[0] = v[1] = v[2] = v[3] = 0.f;
      if (gk < Kmax) {
        const float* g = p + (long long)gk * ld + go;
        if (go + 3 < OUTmax) v = *(const f32x4*)g;
        else { for (int i = 0; i < 4; ++i) if (go + i < OUTmax) v[i] = g[i]; }
      }
#pragma unroll
      for (int i = 0; i < 4; ++i) { float f = v[i]; if (presilu) f = silu_f(f); sm[(o4 + i) * LSTR + k] = f2bf(f); }
    }
  }
}

// ---------------------------------------------------------------- WMMA GEMM
// C[m,n] = act( sum_k A[m,k]*B[k,n] + bias[n] ) (+ res[m,n])
// Batched: z -> (b2 = z/inner, b1 = z%inner); base = b2*s?2 + b1*s?1.
template <int BM, int BN, int WM, int WN>
__global__ __launch_bounds__((BM / WM) * (BN / WN) * 32)
void gemm_bf16(const float* __restrict__ A, const float* __restrict__ Bm,
               const float* __restrict__ bias, const float* __restrict__ res,
               float* __restrict__ C,
               int M, int N, int K, int lda, int ldb, int ldc,
               long long sA1, long long sA2, long long sB1, long long sB2,
               long long sC1, long long sC2, int inner, int flags) {
  const int KT = 32, LSTR = 40;
  const int NWAVES = (BM / WM) * (BN / WN);
  const int NTHR = NWAVES * 32;
  const int MT = WM / 16, NT = WN / 16;

  __shared__ unsigned short sA[BM * 40];
  __shared__ unsigned short sB[BN * 40];

  const int tid = threadIdx.x;
  const int wid = tid >> 5, lane = tid & 31;
  const int wcols = BN / WN;
  const int wm0 = (wid / wcols) * WM;
  const int wn0 = (wid % wcols) * WN;

  const int z = blockIdx.z;
  const int b1 = z % inner, b2 = z / inner;
  const float* Ap = A + (long long)b2 * sA2 + (long long)b1 * sA1;
  const float* Bp = Bm + (long long)b2 * sB2 + (long long)b1 * sB1;
  const long long cbase = (long long)b2 * sC2 + (long long)b1 * sC1;

  const int m0 = blockIdx.y * BM;
  const int n0 = blockIdx.x * BN;

  const bool contigA = !(flags & FLAG_TA);   // A contiguous along K
  const bool contigB =  (flags & FLAG_TB);   // B^T contiguous along K
  const bool psilu   =  (flags & FLAG_PSILU);

  floatx8 acc[MT][NT];
#pragma unroll
  for (int i = 0; i < MT; ++i)
#pragma unroll
    for (int j = 0; j < NT; ++j)
#pragma unroll
      for (int r = 0; r < 8; ++r) acc[i][j][r] = 0.f;

  const int lrow = lane & 15;
  const int kbA = (lane < 16) ? 0 : 8;    // A: chunks at +0/+16 from kbA
  const int kbB = (lane < 16) ? 0 : 8;    // B: chunks at +0/+16 from kbB

  for (int k0 = 0; k0 < K; k0 += KT) {
    stage_tile<BM, (BM / WM) * (BN / WN) * 32>(sA, Ap, lda, contigA, m0, k0, M, K, tid, psilu);
    stage_tile<BN, (BM / WM) * (BN / WN) * 32>(sB, Bp, ldb, contigB, n0, k0, N, K, tid, false);
    __syncthreads();

    FragBF af[MT], bf[NT];
#pragma unroll
    for (int i = 0; i < MT; ++i) {
      const unsigned short* p = sA + (wm0 + 16 * i + lrow) * LSTR + kbA;
      af[i].q[0] = *(const u32x4*)p;
      af[i].q[1] = *(const u32x4*)(p + 16);
    }
#pragma unroll
    for (int j = 0; j < NT; ++j) {
      const unsigned short* p = sB + (wn0 + 16 * j + lrow) * LSTR + kbB;
      bf[j].q[0] = *(const u32x4*)p;
      bf[j].q[1] = *(const u32x4*)(p + 16);
    }
#pragma unroll
    for (int i = 0; i < MT; ++i)
#pragma unroll
      for (int j = 0; j < NT; ++j)
        acc[i][j] = __builtin_amdgcn_wmma_f32_16x16x32_bf16(
            false, af[i].v, false, bf[j].v, (short)0, acc[i][j], false, false);
    __syncthreads();
  }

  const bool gelu = flags & FLAG_GELU;
  const bool silu = flags & FLAG_SILU;
  const int mofs = (lane & 16) ? 8 : 0;
#pragma unroll
  for (int i = 0; i < MT; ++i)
#pragma unroll
    for (int j = 0; j < NT; ++j)
#pragma unroll
      for (int r = 0; r < 8; ++r) {
        int m = m0 + wm0 + 16 * i + r + mofs;
        int n = n0 + wn0 + 16 * j + lrow;
        if (m < M && n < N) {
          float v = acc[i][j][r];
          if (bias) v += bias[n];
          if (gelu) v = gelu_f(v);
          else if (silu) v = silu_f(v);
          long long idx = cbase + (long long)m * ldc + n;
          if (res) v += res[idx];
          C[idx] = v;
        }
      }
}

// ---------------------------------------------------------------- elementwise
__global__ __launch_bounds__(256) void ln_kernel(const float* __restrict__ in,
                                                 const float* __restrict__ g,
                                                 const float* __restrict__ b,
                                                 float* __restrict__ out, int D) {
  const long long row = blockIdx.x;
  const float* x = in + row * D;
  float s = 0.f, s2 = 0.f;
  for (int c = threadIdx.x; c < D; c += 256) { float v = x[c]; s += v; s2 += v * v; }
  __shared__ float r1[256], r2[256];
  int t = threadIdx.x;
  r1[t] = s; r2[t] = s2; __syncthreads();
  for (int st = 128; st > 0; st >>= 1) {
    if (t < st) { r1[t] += r1[t + st]; r2[t] += r2[t + st]; }
    __syncthreads();
  }
  float mean = r1[0] / D, var = r2[0] / D - (r1[0] / D) * (r1[0] / D);
  float inv = rsqrtf(var + 1e-5f);
  for (int c = t; c < D; c += 256) out[row * D + c] = (x[c] - mean) * inv * g[c] + b[c];
}

// FiLM-modulated LN followed by SiLU:  out = silu( ln(x)*(1+scale_b) + shift_b )
__global__ __launch_bounds__(256) void film_ln_silu_kernel(const float* __restrict__ in,
                                                           const float* __restrict__ eo,
                                                           const float* __restrict__ g,
                                                           const float* __restrict__ b,
                                                           float* __restrict__ out,
                                                           int T, int D) {
  const long long row = blockIdx.x;
  const int bi = (int)(row / T);
  const float* x = in + row * D;
  float s = 0.f, s2 = 0.f;
  for (int c = threadIdx.x; c < D; c += 256) { float v = x[c]; s += v; s2 += v * v; }
  __shared__ float r1[256], r2[256];
  int t = threadIdx.x;
  r1[t] = s; r2[t] = s2; __syncthreads();
  for (int st = 128; st > 0; st >>= 1) {
    if (t < st) { r1[t] += r1[t + st]; r2[t] += r2[t + st]; }
    __syncthreads();
  }
  float mean = r1[0] / D, var = r2[0] / D - (r1[0] / D) * (r1[0] / D);
  float inv = rsqrtf(var + 1e-5f);
  const float* sc = eo + (long long)bi * 2 * D;
  const float* sh = sc + D;
  for (int c = t; c < D; c += 256) {
    float v = (x[c] - mean) * inv * g[c] + b[c];
    float h = v * (1.f + sc[c]) + sh[c];
    out[row * D + c] = silu_f(h);
  }
}

// softmax over trailing 64-element segment; one wave32 per row
__global__ __launch_bounds__(256) void softmax64_kernel(const float* __restrict__ in,
                                                        float* __restrict__ out,
                                                        long long rows) {
  int w = threadIdx.x >> 5, lane = threadIdx.x & 31;
  long long row = (long long)blockIdx.x * 8 + w;
  if (row >= rows) return;
  const float* p = in + row * 64;
  float v0 = p[lane], v1 = p[lane + 32];
  float m = fmaxf(v0, v1);
  for (int o = 16; o > 0; o >>= 1) m = fmaxf(m, __shfl_xor(m, o, 32));
  float e0 = expf(v0 - m), e1 = expf(v1 - m);
  float s = e0 + e1;
  for (int o = 16; o > 0; o >>= 1) s += __shfl_xor(s, o, 32);
  out[row * 64 + lane] = e0 / s;
  out[row * 64 + lane + 32] = e1 / s;
}

// column softmax over axis 1 of (B, L, D); one block per (b, d) column
__global__ __launch_bounds__(256) void softmax_col_kernel(const float* __restrict__ in,
                                                          float* __restrict__ out,
                                                          int L, int D) {
  const int col = blockIdx.x;
  const int b = col / D, c = col % D;
  const long long base = (long long)b * L * D + c;
  float vals[8];
  int t = threadIdx.x, n = 0;
  float m = -3.0e38f;
  for (int l = t; l < L; l += 256) { float v = in[base + (long long)l * D]; vals[n++] = v; m = fmaxf(m, v); }
  __shared__ float red[256];
  red[t] = m; __syncthreads();
  for (int st = 128; st > 0; st >>= 1) { if (t < st) red[t] = fmaxf(red[t], red[t + st]); __syncthreads(); }
  m = red[0]; __syncthreads();
  float s = 0.f;
  for (int i = 0; i < n; ++i) { vals[i] = expf(vals[i] - m); s += vals[i]; }
  red[t] = s; __syncthreads();
  for (int st = 128; st > 0; st >>= 1) { if (t < st) red[t] += red[t + st]; __syncthreads(); }
  float invs = 1.f / red[0];
  n = 0;
  for (int l = t; l < L; l += 256) out[base + (long long)l * D] = vals[n++] * invs;
}

// ---------------------------------------------------------------- host wrapper
static void gemm(hipStream_t st, const float* A, const float* Bm, const float* bias,
                 const float* res, float* C, int M, int N, int K, int lda, int ldb, int ldc,
                 long long sA1, long long sA2, long long sB1, long long sB2,
                 long long sC1, long long sC2, int inner, int nbatch, int flags, bool big) {
  if (big) {
    dim3 g((N + 127) / 128, (M + 127) / 128, nbatch);
    gemm_bf16<128, 128, 64, 32><<<g, dim3(256), 0, st>>>(A, Bm, bias, res, C, M, N, K, lda, ldb, ldc,
                                                         sA1, sA2, sB1, sB2, sC1, sC2, inner, flags);
  } else {
    dim3 g((N + 63) / 64, (M + 63) / 64, nbatch);
    gemm_bf16<64, 64, 32, 32><<<g, dim3(128), 0, st>>>(A, Bm, bias, res, C, M, N, K, lda, ldb, ldc,
                                                       sA1, sA2, sB1, sB2, sC1, sC2, inner, flags);
  }
}

extern "C" void kernel_launch(void* const* d_in, const int* in_sizes, int n_in,
                              void* d_out, int out_size, void* d_ws, size_t ws_size,
                              hipStream_t stream) {
  (void)in_sizes; (void)n_in; (void)out_size; (void)ws_size;
  // input order (setup_inputs dict, depth-first):
  // 0 x  1 xf  2 emb  3 tgt_mask  4 memory_mask
  // 5 sa.ng 6 sa.nb 7 sa.qw 8 sa.qb 9 sa.kw 10 sa.kb 11 sa.vw 12 sa.vb
  // 13 sa.sc.ew 14 sa.sc.eb 15 sa.sc.ng 16 sa.sc.nb 17 sa.sc.ow 18 sa.sc.ob
  // 19 ca.ng 20 ca.nb 21 ca.tng 22 ca.tnb 23 ca.qw 24 ca.qb 25 ca.kw 26 ca.kb
  // 27 ca.vw 28 ca.vb 29 ca.sc.ew 30 ca.sc.eb 31 ca.sc.ng 32 ca.sc.nb 33 ca.sc.ow 34 ca.sc.ob
  // 35 ffn.w1 36 ffn.b1 37 ffn.w2 38 ffn.b2
  // 39 ffn.sc.ew 40 ffn.sc.eb 41 ffn.sc.ng 42 ffn.sc.nb 43 ffn.sc.ow 44 ffn.sc.ob
  const float* x   = (const float*)d_in[0];
  const float* xf  = (const float*)d_in[1];
  const float* emb = (const float*)d_in[2];
  const float* tgt = (const float*)d_in[3];
  const float* mm  = (const float*)d_in[4];
  #define P(i) ((const float*)d_in[i])

  float* ws = (float*)d_ws;
  const unsigned long long BTD = 16777216ull;        // B*T*D
  float* XN   = ws + 0 * BTD;
  float* Q    = ws + 1 * BTD;    // FFN hidden reuses [1*BTD, 5*BTD)
  float* Kb   = ws + 2 * BTD;
  float* V    = ws + 3 * BTD;
  float* VM   = ws + 4 * BTD;
  float* Y    = ws + 5 * BTD;
  float* Hb   = ws + 6 * BTD;
  float* Xc   = ws + 7 * BTD;
  float* R7   = ws + 8 * BTD;
  float* OUTb = ws + 9 * BTD;    // B*N*H*T = 4*BTD -> [9*BTD, 13*BTD)
  float* ATTN = ws + 13 * BTD;   // B*H*d*d = 524288
  float* EO   = ATTN + 524288;   // B*2D = 16384
  float* XFN  = EO + 16384;      // B*N*D = 2097152
  float* FFNH = Q;               // 16384 x 4096
  float* out  = (float*)d_out;

  const long long TD = 2097152, TT = 4194304, ND = 262144;   // T*D, T*T, N*D
  const long long HT = 32768;                                // H*T
  const long long OUTB = 8388608;                            // N*H*T per b
  const long long R7B = 2097152;                             // d*H*T per b

  // ===================== self-attention =====================
  ln_kernel<<<16384, 256, 0, stream>>>(x, P(5), P(6), XN, D_);
  gemm(stream, XN, P(7), P(8), nullptr, Q, 16384, D_, D_, D_, D_, D_, 0,0,0,0,0,0, 1,1, 0, true);
  softmax64_kernel<<<32768, 256, 0, stream>>>(Q, Q, 262144ll);
  gemm(stream, XN, P(9), P(10), nullptr, Kb, 16384, D_, D_, D_, D_, D_, 0,0,0,0,0,0, 1,1, 0, true);
  softmax_col_kernel<<<8192, 256, 0, stream>>>(Kb, Kb, T_, D_);
  gemm(stream, XN, P(11), P(12), nullptr, V, 16384, D_, D_, D_, D_, D_, 0,0,0,0,0,0, 1,1, 0, true);
  // v = tgt_mask @ v   (per b: TxT @ TxD)
  gemm(stream, tgt, V, nullptr, nullptr, VM, T_, D_, T_, T_, D_, D_, 0,TT, 0,TD, 0,TD, 1,B_, 0, true);
  // attn = k^T @ v   (per b,h: dxT @ Txd)
  gemm(stream, Kb, VM, nullptr, nullptr, ATTN, DH_, DH_, T_, D_, D_, DH_,
       DH_,TD, DH_,TD, 4096,65536, H_, B_*H_, FLAG_TA, false);
  // y = q @ attn   (per b,h: Txd @ dxd)
  gemm(stream, Q, ATTN, nullptr, nullptr, Y, T_, DH_, DH_, D_, DH_, D_,
       DH_,TD, 4096,65536, DH_,TD, H_, B_*H_, 0, false);
  // scaling converter
  gemm(stream, emb, P(13), P(14), nullptr, EO, B_, 2*D_, TE_, TE_, 2*D_, 2*D_,
       0,0,0,0,0,0, 1,1, FLAG_PSILU, false);
  film_ln_silu_kernel<<<16384, 256, 0, stream>>>(Y, EO, P(15), P(16), Hb, T_, D_);
  gemm(stream, Hb, P(17), P(18), x, Xc, 16384, D_, D_, D_, D_, D_, 0,0,0,0,0,0, 1,1, 0, true);

  // ===================== cross-attention =====================
  ln_kernel<<<16384, 256, 0, stream>>>(Xc, P(19), P(20), XN, D_);
  ln_kernel<<<2048, 256, 0, stream>>>(xf, P(21), P(22), XFN, D_);
  gemm(stream, XN, P(23), P(24), nullptr, Q, 16384, D_, D_, D_, D_, D_, 0,0,0,0,0,0, 1,1, 0, true);
  softmax64_kernel<<<32768, 256, 0, stream>>>(Q, Q, 262144ll);
  gemm(stream, XFN, P(25), P(26), nullptr, Kb, 2048, D_, D_, D_, D_, D_, 0,0,0,0,0,0, 1,1, 0, true);
  softmax_col_kernel<<<8192, 256, 0, stream>>>(Kb, Kb, NN_, D_);
  gemm(stream, XFN, P(27), P(28), nullptr, V, 2048, D_, D_, D_, D_, D_, 0,0,0,0,0,0, 1,1, 0, true);
  // v = memory_mask @ v   (per b: TxN @ NxD) -> (B,T,D) viewed as (B,H,d,T)
  gemm(stream, mm, V, nullptr, nullptr, VM, T_, D_, NN_, NN_, D_, D_, 0,(long long)T_*NN_, 0,ND, 0,TD, 1,B_, 0, true);
  // out = k @ v4   (per b,h: Nxd @ dxT) -> (B,N,H,T)
  gemm(stream, Kb, VM, nullptr, nullptr, OUTb, NN_, T_, DH_, D_, T_, HT,
       DH_,ND, (long long)DH_*T_,TD, T_,OUTB, H_, B_*H_, 0, true);
  // r7 = k^T @ out   (per b,h: dxN @ NxT) -> (B,d,H,T)
  gemm(stream, Kb, OUTb, nullptr, nullptr, R7, DH_, T_, NN_, D_, HT, HT,
       DH_,ND, T_,OUTB, T_,R7B, H_, B_*H_, FLAG_TA, false);
  // attn = q^T @ out4   (per b,h: dxT @ Txd), out4[t][l] = r7[b,l,h,t]
  gemm(stream, Q, R7, nullptr, nullptr, ATTN, DH_, DH_, T_, D_, HT, DH_,
       DH_,TD, T_,R7B, 4096,65536, H_, B_*H_, FLAG_TA | FLAG_TB, false);
  // y = q @ attn
  gemm(stream, Q, ATTN, nullptr, nullptr, Y, T_, DH_, DH_, D_, DH_, D_,
       DH_,TD, 4096,65536, DH_,TD, H_, B_*H_, 0, false);
  gemm(stream, emb, P(29), P(30), nullptr, EO, B_, 2*D_, TE_, TE_, 2*D_, 2*D_,
       0,0,0,0,0,0, 1,1, FLAG_PSILU, false);
  film_ln_silu_kernel<<<16384, 256, 0, stream>>>(Y, EO, P(31), P(32), Hb, T_, D_);
  gemm(stream, Hb, P(33), P(34), Xc, Xc, 16384, D_, D_, D_, D_, D_, 0,0,0,0,0,0, 1,1, 0, true);

  // ===================== FFN =====================
  gemm(stream, Xc, P(35), P(36), nullptr, FFNH, 16384, FD_, D_, D_, FD_, FD_,
       0,0,0,0,0,0, 1,1, FLAG_GELU, true);
  gemm(stream, FFNH, P(37), P(38), nullptr, Y, 16384, D_, FD_, FD_, D_, D_,
       0,0,0,0,0,0, 1,1, 0, true);
  gemm(stream, emb, P(39), P(40), nullptr, EO, B_, 2*D_, TE_, TE_, 2*D_, 2*D_,
       0,0,0,0,0,0, 1,1, FLAG_PSILU, false);
  film_ln_silu_kernel<<<16384, 256, 0, stream>>>(Y, EO, P(41), P(42), Hb, T_, D_);
  gemm(stream, Hb, P(43), P(44), Xc, out, 16384, D_, D_, D_, D_, D_, 0,0,0,0,0,0, 1,1, 0, true);
  #undef P
}
